// GatedSpatialLinearAttention_50861002719378
// MI455X (gfx1250) — compile-verified, matrix operand order checked by reading, not run on hardware
//
#include <hip/hip_runtime.h>
#include <hip/hip_bf16.h>

// ---------------------------------------------------------------------------
// GatedSpatialLinearAttention for MI455X (gfx1250), wave32 + WMMA bf16.
// x:(2,256,16,64,64) f32. Pipeline:
//   k_cvt  : weights f32 -> bf16
//   k_xT   : x -> xbT[frame][s][c] bf16 (channel-innermost, one-time transpose)
//   k1_kv  : per (frame,head): kv[64x64], k_sum[64]   (WMMA bf16)
//   k2_hid : per (frame,head): hidT[frame][s][ch] bf16 (WMMA bf16)
//   k3_out : out = Wo @ hid + bo                       (WMMA bf16)
// LDS staging uses GLOBAL_LOAD_ASYNC_TO_LDS_B128 (ASYNCcnt) and the Tensor
// Data Mover (TENSORcnt) where the toolchain exposes them; falls back to
// b128 vector copies otherwise.
// ---------------------------------------------------------------------------

typedef __attribute__((ext_vector_type(16))) __bf16 v16bf;
typedef __attribute__((ext_vector_type(8)))  __bf16 v8bf;
typedef __attribute__((ext_vector_type(8)))  float  v8f;
typedef __attribute__((ext_vector_type(4)))  unsigned int u32x4;
typedef __attribute__((ext_vector_type(8)))  int i32x8;
typedef __attribute__((ext_vector_type(4)))  int i32x4;
typedef int v4i __attribute__((vector_size(16)));  // matches async builtin param

#define DIMC   256
#define FRAMES 16
#define BFR    32      // b * f
#define HEADS  8
#define DHEAD  64
#define HIDDEN 512
#define HW     64
#define NSPAT  4096
#define SCALE_Q 0.125f // 64^-0.5

// ---- CDNA5 feature detection (device pass only) ---------------------------
#if defined(__gfx1250__) && __has_builtin(__builtin_amdgcn_global_load_async_to_lds_b128) && \
    __has_builtin(__builtin_amdgcn_s_wait_asynccnt)
#define HAS_ASYNC 1
#else
#define HAS_ASYNC 0
#endif

#if defined(__gfx1250__) && __has_builtin(__builtin_amdgcn_tensor_load_to_lds) && \
    __has_builtin(__builtin_amdgcn_s_wait_tensorcnt)
#define HAS_TDM 1
#else
#define HAS_TDM 0
#endif

// 16B global -> LDS copy, async when available
__device__ __forceinline__ void copy16_async(const v8bf* g, v8bf* l) {
#if HAS_ASYNC
  __builtin_amdgcn_global_load_async_to_lds_b128((v4i*)g, (v4i*)l, 0, 0);
#else
  *l = *g;
#endif
}
__device__ __forceinline__ void async_join() {
#if HAS_ASYNC
  __builtin_amdgcn_s_wait_asynccnt(0);
#endif
}

#if HAS_TDM
// 1-D contiguous TDM load: nelem bf16 elements global -> LDS (single wave issues).
__device__ __forceinline__ void tdm_load_1d(const void* gsrc, void* lds, unsigned nelem) {
  const unsigned long long ga = (unsigned long long)(uintptr_t)gsrc;
  const unsigned la =
      (unsigned)(uintptr_t)(__attribute__((address_space(3))) char*)lds;
  u32x4 g0;
  g0[0] = 1u;                                   // count=1, user descriptor
  g0[1] = la;                                   // lds_addr
  g0[2] = (unsigned)ga;                         // global_addr[31:0]
  g0[3] = (unsigned)(ga >> 32) | (2u << 30);    // global_addr[56:32] | type=2
  i32x8 g1;
  g1[0] = (int)(1u << 16);                      // workgroup_mask=0, data_size=2B
  g1[1] = (int)((nelem & 0xFFFFu) << 16);       // tensor_dim0[15:0]
  g1[2] = (int)((nelem >> 16) | (1u << 16));    // tensor_dim0[31:16] | tensor_dim1=1
  g1[3] = (int)((nelem & 0xFFFFu) << 16);       // tile_dim0
  g1[4] = 0;                                    // tile_dim1=0 (1-D), tile_dim2=0
  g1[5] = (int)nelem;                           // tensor_dim0_stride[31:0]
  g1[6] = 0;
  g1[7] = 0;
  i32x4 g2 = {0, 0, 0, 0};
  i32x4 g3 = {0, 0, 0, 0};
#if __clang_major__ >= 23
  i32x8 g4 = {0, 0, 0, 0, 0, 0, 0, 0};
  __builtin_amdgcn_tensor_load_to_lds(g0, g1, g2, g3, g4, 0);
#else
  __builtin_amdgcn_tensor_load_to_lds(g0, g1, g2, g3, 0);
#endif
}
#endif  // HAS_TDM

__device__ __forceinline__ v8f wmma_bf16(v16bf a, v16bf b, v8f c) {
  return __builtin_amdgcn_wmma_f32_16x16x32_bf16(
      false, a, false, b, (short)0, c, false, false);
}

// Fragment load per CDNA5 ISA 7.12.2 (16-bit 16x32 layout, half-wave K split).
// src is [rows][K] with K contiguous; rows = M for A, rows = N for B.
// Requires ld % 8 == 0 and k0 % 32 == 0 -> two aligned ds_load_b128.
__device__ __forceinline__ v16bf frag_ld(const __bf16* src, int ld, int r0, int k0) {
  const int lane = threadIdx.x & 31;
  const __bf16* p = src + (r0 + (lane & 15)) * ld + k0 + ((lane >> 4) << 3);
  const v8bf lo = *(const v8bf*)p;
  const v8bf hi = *(const v8bf*)(p + 16);
  v16bf out;
#pragma unroll
  for (int i = 0; i < 8; ++i) { out[i] = lo[i]; out[8 + i] = hi[i]; }
  return out;
}

// C/D 16x16 f32: lane n = L&15, VGPR r -> M = r + 8*(L>>4)
__device__ __forceinline__ void store_c(float* dst, int ld, int row0, int col0, v8f c) {
  const int lane = threadIdx.x & 31;
  const int n = lane & 15, half = lane >> 4;
#pragma unroll
  for (int r = 0; r < 8; ++r)
    dst[(row0 + r + 8 * half) * ld + col0 + n] = c[r];
}

__device__ __forceinline__ void store_c_bf(__bf16* dst, int ld, int row0, int col0, v8f c) {
  const int lane = threadIdx.x & 31;
  const int n = lane & 15, half = lane >> 4;
#pragma unroll
  for (int r = 0; r < 8; ++r)
    dst[(row0 + r + 8 * half) * ld + col0 + n] = (__bf16)c[r];
}

// ---------------------------------------------------------------------------
// Elementwise f32 -> bf16 (weights)
// ---------------------------------------------------------------------------
__global__ __launch_bounds__(256) void k_cvt(const float* __restrict__ in,
                                             __bf16* __restrict__ outp, int n) {
  const int i = blockIdx.x * 256 + threadIdx.x;
  if (i < n) outp[i] = (__bf16)in[i];
}

// ---------------------------------------------------------------------------
// x[b][c][f][s] f32 -> xbT[bf][s][c] bf16  (LDS tile transpose, padded)
// ---------------------------------------------------------------------------
__global__ __launch_bounds__(256) void k_xT(const float* __restrict__ x,
                                            __bf16* __restrict__ xbT) {
  __shared__ __bf16 sT[DIMC * 65];  // [c][s], ld 65 to dodge bank conflicts
  const int tid   = threadIdx.x;
  const int bf    = blockIdx.x >> 6;
  const int chunk = blockIdx.x & 63;
  const int b = bf >> 4, f = bf & 15;
  const size_t xbase = (((size_t)b * DIMC) * FRAMES + f) * NSPAT + chunk * 64;

  for (int i = tid; i < DIMC * 64; i += 256) {
    const int c = i >> 6, s = i & 63;
    sT[c * 65 + s] = (__bf16)x[xbase + (size_t)c * FRAMES * NSPAT + s];
  }
  __syncthreads();
  for (int i = tid; i < DIMC * 64; i += 256) {
    const int s = i >> 8, c = i & 255;
    xbT[((size_t)bf * NSPAT + chunk * 64 + s) * DIMC + c] = sT[c * 65 + s];
  }
}

// ---------------------------------------------------------------------------
// Kernel 1: per (frame, head) compute kv[64x64] and k_sum[64].
//   k = elu(softmax_w(Wk_head @ X)) + 1, one h-row (64 w) per step.
// ---------------------------------------------------------------------------
__global__ __launch_bounds__(256) void k1_kv(
    const __bf16* __restrict__ xbT, const __bf16* __restrict__ Wkb,
    const __bf16* __restrict__ Wvb, float* __restrict__ kv_out,
    float* __restrict__ ksum_out) {
  extern __shared__ char smem[];
  float*  sK  = (float*)smem;                 // 64x64 f32 (pre-softmax K)
  __bf16* sKb = (__bf16*)(sK + 4096);         // 64x64 [d][s]
  __bf16* sVb = sKb + 4096;                   // 64x64 [e][s]
  __bf16* sWk = sVb + 4096;                   // 64x256 [d][c]
  __bf16* sWv = sWk + DHEAD * DIMC;           // 64x256
  __bf16* sXT = sWv + DHEAD * DIMC;           // 64x256 [s][c]

  const int tid  = threadIdx.x;
  const int wave = tid >> 5;
  const int bf   = blockIdx.x >> 3;
  const int head = blockIdx.x & 7;

  {  // weights (bf16, contiguous) -> async to LDS
    const v8bf* gk = (const v8bf*)&Wkb[head * DHEAD * DIMC];
    const v8bf* gv = (const v8bf*)&Wvb[head * DHEAD * DIMC];
    v8bf* lk = (v8bf*)sWk;
    v8bf* lv = (v8bf*)sWv;
    for (int i = tid; i < DHEAD * DIMC / 8; i += 256) {
      copy16_async(gk + i, lk + i);
      copy16_async(gv + i, lv + i);
    }
  }

  v8f acc0 = {}, acc1 = {};  // this wave's 2 kv tiles (16 tiles over 8 waves)
  float ksum_acc = 0.f;

  for (int row = 0; row < HW; ++row) {
    __syncthreads();
    {  // X tile: 64 spatial rows x 256 channels, contiguous async copy
      const v8bf* gsrc = (const v8bf*)&xbT[((size_t)bf * NSPAT + row * HW) * DIMC];
      v8bf* ldst = (v8bf*)sXT;
      for (int i = tid; i < HW * DIMC / 8; i += 256) copy16_async(gsrc + i, ldst + i);
    }
    if (row + 1 < HW)  // global_prefetch_b8 of next row tile
      __builtin_prefetch(&xbT[((size_t)bf * NSPAT + (row + 1) * HW) * DIMC + tid * 64], 0, 1);
    async_join();
    __syncthreads();

    // K and V 64x64 tiles: 32 (16x16)-tiles over 8 waves
    for (int t = wave; t < 32; t += 8) {
      const __bf16* W = (t < 16) ? sWk : sWv;
      const int tt = t & 15, mt = tt >> 2, nt = tt & 3;
      v8f c = {};
#pragma unroll
      for (int k0 = 0; k0 < DIMC; k0 += 32)
        c = wmma_bf16(frag_ld(W, DIMC, mt * 16, k0),
                      frag_ld(sXT, DIMC, nt * 16, k0), c);
      if (t < 16) store_c(sK, HW, mt * 16, nt * 16, c);
      else        store_c_bf(sVb, HW, mt * 16, nt * 16, c);
    }
    __syncthreads();

    // softmax over w per channel row, elu+1, write bf16 row, track ksum
    if (tid < DHEAD) {
      float* r = sK + tid * HW;
      __bf16* rb = sKb + tid * HW;
      float mx = r[0];
      for (int j = 1; j < HW; ++j) mx = fmaxf(mx, r[j]);
      float ssum = 0.f;
      for (int j = 0; j < HW; ++j) { float e = __expf(r[j] - mx); r[j] = e; ssum += e; }
      const float inv = 1.f / ssum;
      float rs = 0.f;
      for (int j = 0; j < HW; ++j) {
        float p = r[j] * inv;
        float feat = (p > 0.f ? p : expm1f(p)) + 1.f;
        rb[j] = (__bf16)feat; rs += feat;
      }
      ksum_acc += rs;
    }
    __syncthreads();

    // kv[d,e] += sum_s Kfeat[d,s] * V[e,s]
#pragma unroll
    for (int j = 0; j < 2; ++j) {
      const int t = wave * 2 + j, mt = t >> 2, nt = t & 3;
      v8f* acc = j ? &acc1 : &acc0;
#pragma unroll
      for (int k0 = 0; k0 < HW; k0 += 32)
        *acc = wmma_bf16(frag_ld(sKb, HW, mt * 16, k0),
                         frag_ld(sVb, HW, nt * 16, k0), *acc);
    }
  }

  float* kvp = kv_out + (size_t)blockIdx.x * DHEAD * DHEAD;  // [d][e]
#pragma unroll
  for (int j = 0; j < 2; ++j) {
    const int t = wave * 2 + j, mt = t >> 2, nt = t & 3;
    store_c(kvp, DHEAD, mt * 16, nt * 16, j ? acc1 : acc0);
  }
  if (tid < DHEAD) ksum_out[(size_t)blockIdx.x * DHEAD + tid] = ksum_acc;
}

// ---------------------------------------------------------------------------
// Kernel 2: per (frame, head): hidT[s][ch] = SCALE * kv^T @ q_feat * z * g
// ---------------------------------------------------------------------------
__global__ __launch_bounds__(256) void k2_hid(
    const __bf16* __restrict__ xbT, const __bf16* __restrict__ Wqb,
    const __bf16* __restrict__ Wgb, const float* __restrict__ bg,
    const float* __restrict__ kv_in, const float* __restrict__ ksum_in,
    __bf16* __restrict__ hidT) {
  extern __shared__ char smem[];
  float*  sQ   = (float*)smem;        // 64x64 [d][h] f32
  float*  sG   = sQ + 4096;           // 64x64 [e][h] f32
  float*  sZ   = sG + 4096;           // 64
  float*  sKs  = sZ + 64;             // 64
  float*  sBg  = sKs + 64;            // 64
  __bf16* sKVt = (__bf16*)(sBg + 64); // 64x64 [e][d]
  __bf16* sQb  = sKVt + 4096;         // 64x64 [h][d]
  __bf16* sWq  = sQb + 4096;          // 64x256 [d][c]
  __bf16* sWg  = sWq + DHEAD * DIMC;  // 64x256
  __bf16* sXT  = sWg + DHEAD * DIMC;  // 64x256 [h][c]

  const int tid  = threadIdx.x;
  const int wave = tid >> 5;
  const int bf   = blockIdx.x >> 3;
  const int head = blockIdx.x & 7;

  {
    const v8bf* gq = (const v8bf*)&Wqb[head * DHEAD * DIMC];
    const v8bf* gg = (const v8bf*)&Wgb[head * DHEAD * DIMC];
    v8bf* lq = (v8bf*)sWq;
    v8bf* lg = (v8bf*)sWg;
    for (int i = tid; i < DHEAD * DIMC / 8; i += 256) {
      copy16_async(gq + i, lq + i);
      copy16_async(gg + i, lg + i);
    }
  }
  for (int i = tid; i < 4096; i += 256) {   // kv [d][e] f32 -> [e][d] bf16
    const int d = i >> 6, e = i & 63;
    sKVt[e * DHEAD + d] = (__bf16)kv_in[(size_t)blockIdx.x * 4096 + i];
  }
  if (tid < DHEAD) {
    sKs[tid] = ksum_in[(size_t)blockIdx.x * DHEAD + tid];
    sBg[tid] = bg[head * DHEAD + tid];
  }

  for (int w = 0; w < HW; ++w) {
    __syncthreads();
    {  // X columns: rows h -> xbT[bf][h*64+w][c], 512B contiguous per row
      v8bf* ldst = (v8bf*)sXT;
      for (int i = tid; i < HW * (DIMC / 8); i += 256) {
        const int h = i >> 5, cv = i & 31;
        copy16_async((const v8bf*)&xbT[((size_t)bf * NSPAT + h * HW + w) * DIMC + cv * 8],
                     ldst + i);
      }
    }
    if (w + 1 < HW && tid < HW)
      __builtin_prefetch(&xbT[((size_t)bf * NSPAT + tid * HW + w + 1) * DIMC], 0, 1);
    async_join();
    __syncthreads();

    // q and g 64x64 tiles (rows = channel, cols = h)
    for (int t = wave; t < 32; t += 8) {
      const __bf16* W = (t < 16) ? sWq : sWg;
      float* dst = (t < 16) ? sQ : sG;
      const int tt = t & 15, mt = tt >> 2, nt = tt & 3;
      v8f c = {};
#pragma unroll
      for (int k0 = 0; k0 < DIMC; k0 += 32)
        c = wmma_bf16(frag_ld(W, DIMC, mt * 16, k0),
                      frag_ld(sXT, DIMC, nt * 16, k0), c);
      store_c(dst, HW, mt * 16, nt * 16, c);
    }
    __syncthreads();

    // softmax over h per channel, elu+1 -> sQ (f32, for z) and sQb [h][d]
    if (tid < DHEAD) {
      float* r = sQ + tid * HW;
      float mx = r[0];
      for (int j = 1; j < HW; ++j) mx = fmaxf(mx, r[j]);
      float ssum = 0.f;
      for (int j = 0; j < HW; ++j) { float e = __expf(r[j] - mx); r[j] = e; ssum += e; }
      const float inv = 1.f / ssum;
      for (int j = 0; j < HW; ++j) {
        float p = r[j] * inv;
        float feat = (p > 0.f ? p : expm1f(p)) + 1.f;
        r[j] = feat;
        sQb[j * DHEAD + tid] = (__bf16)feat;  // lane-contiguous transposed write
      }
    }
    for (int i = tid; i < 4096; i += 256) {  // gate = sigmoid(g + bg)
      const int d = i >> 6;
      sG[i] = 1.f / (1.f + __expf(-(sG[i] + sBg[d])));
    }
    __syncthreads();
    if (tid < HW) {  // z[h] = 1 / max(q_feat[:,h] . ksum, 1e-6)
      float zz = 0.f;
      for (int d = 0; d < DHEAD; ++d) zz += sQ[d * HW + tid] * sKs[d];
      sZ[tid] = 1.f / fmaxf(zz, 1e-6f);
    }
    __syncthreads();

    // hid[e,h] = SCALE * (kv^T @ q_feat)[e,h] * z[h] * g[e,h]; packed b128 store
#pragma unroll
    for (int j = 0; j < 2; ++j) {
      const int t = wave * 2 + j, mt = t >> 2, nt = t & 3;
      v8f c = {};
#pragma unroll
      for (int k0 = 0; k0 < DHEAD; k0 += 32)
        c = wmma_bf16(frag_ld(sKVt, DHEAD, mt * 16, k0),
                      frag_ld(sQb, DHEAD, nt * 16, k0), c);
      const int lane = tid & 31, n = lane & 15, half = lane >> 4;
      const int hh = nt * 16 + n;
      const int e0 = mt * 16 + 8 * half;  // 8 consecutive e-channels per lane
      const float zs = SCALE_Q * sZ[hh];
      v8bf pk;
#pragma unroll
      for (int r = 0; r < 8; ++r)
        pk[r] = (__bf16)(c[r] * zs * sG[(e0 + r) * HW + hh]);
      *(v8bf*)&hidT[((size_t)bf * NSPAT + hh * HW + w) * HIDDEN + head * DHEAD + e0] = pk;
    }
  }
}

// ---------------------------------------------------------------------------
// Kernel 3: out[c,s] = Wo(256x512) @ hid(512 x 64-chunk) + bo, f32 store.
// hid tile (64KB contiguous) fetched by the Tensor Data Mover when available.
// ---------------------------------------------------------------------------
__global__ __launch_bounds__(256) void k3_out(
    const __bf16* __restrict__ hidT, const __bf16* __restrict__ Wob,
    const float* __restrict__ bo, float* __restrict__ out) {
  extern __shared__ char smem[];
  __bf16* sHT  = (__bf16*)smem;        // 64x512 [s][ch]
  __bf16* sWoS = sHT + HW * HIDDEN;    // 256x32 [c_out][k] slice

  const int tid   = threadIdx.x;
  const int wave  = tid >> 5;
  const int bf    = blockIdx.x >> 6;
  const int chunk = blockIdx.x & 63;
  const int b = bf >> 4, f = bf & 15;

  const __bf16* gH = &hidT[((size_t)bf * NSPAT + chunk * HW) * HIDDEN];
#if HAS_TDM
  if (wave == 0) {  // single-wave TDM issue; EXEC ignored by tensor ops
    tdm_load_1d(gH, sHT, HW * HIDDEN);
    __builtin_amdgcn_s_wait_tensorcnt(0);
  }
#else
  {
    const v8bf* gsrc = (const v8bf*)gH;
    v8bf* ldst = (v8bf*)sHT;
    for (int i = tid; i < HW * HIDDEN / 8; i += 256) copy16_async(gsrc + i, ldst + i);
    async_join();
  }
#endif

  v8f acc[8] = {};  // 64 tiles (16m x 4n) over 8 waves: t = wave*8+j
  for (int k0 = 0; k0 < HIDDEN; k0 += 32) {
    __syncthreads();
    {  // Wo K-slice
      v8bf* ldst = (v8bf*)sWoS;
      for (int i = tid; i < DIMC * 4; i += 256) {
        const int r = i >> 2, kk = i & 3;
        copy16_async((const v8bf*)&Wob[(size_t)r * HIDDEN + k0 + kk * 8], ldst + i);
      }
      async_join();
    }
    __syncthreads();
#pragma unroll
    for (int j = 0; j < 8; ++j) {
      const int t = wave * 8 + j, mt = t >> 2, nt = t & 3;
      acc[j] = wmma_bf16(frag_ld(sWoS, 32, mt * 16, 0),
                         frag_ld(sHT, HIDDEN, nt * 16, k0), acc[j]);
    }
  }

  const int lane = tid & 31, n = lane & 15, half = lane >> 4;
#pragma unroll
  for (int j = 0; j < 8; ++j) {
    const int t = wave * 8 + j, mt = t >> 2, nt = t & 3;
#pragma unroll
    for (int r8 = 0; r8 < 8; ++r8) {
      const int c = mt * 16 + r8 + 8 * half;
      const int s = nt * 16 + n;
      out[(((size_t)b * DIMC + c) * FRAMES + f) * NSPAT + chunk * HW + s] =
          acc[j][r8] + bo[c];
    }
  }
}

// ---------------------------------------------------------------------------
extern "C" void kernel_launch(void* const* d_in, const int* in_sizes, int n_in,
                              void* d_out, int out_size, void* d_ws, size_t ws_size,
                              hipStream_t stream) {
  (void)in_sizes; (void)n_in; (void)out_size; (void)ws_size;
  const float* x  = (const float*)d_in[0];
  const float* Wq = (const float*)d_in[1];
  const float* Wk = (const float*)d_in[2];
  const float* Wv = (const float*)d_in[3];
  const float* Wg = (const float*)d_in[4];
  const float* bg = (const float*)d_in[5];
  const float* Wo = (const float*)d_in[6];
  const float* bo = (const float*)d_in[7];
  float* out = (float*)d_out;

  // workspace: xbT 64MB | Wqb/Wkb/Wvb/Wgb/Wob 1.25MB | hidT 128MB | kv 4MB | ksum 64KB
  __bf16* xbT  = (__bf16*)d_ws;
  __bf16* Wqb  = xbT + (size_t)BFR * NSPAT * DIMC;
  __bf16* Wkb  = Wqb + HIDDEN * DIMC;
  __bf16* Wvb  = Wkb + HIDDEN * DIMC;
  __bf16* Wgb  = Wvb + HIDDEN * DIMC;
  __bf16* Wob  = Wgb + HIDDEN * DIMC;
  __bf16* hidT = Wob + DIMC * HIDDEN;
  float*  kv   = (float*)(hidT + (size_t)BFR * NSPAT * HIDDEN);
  float*  ksum = kv + BFR * HEADS * DHEAD * DHEAD;

  const dim3 blk(256);
  const int nw = HIDDEN * DIMC;
  k_cvt<<<dim3((nw + 255) / 256), blk, 0, stream>>>(Wq, Wqb, nw);
  k_cvt<<<dim3((nw + 255) / 256), blk, 0, stream>>>(Wk, Wkb, nw);
  k_cvt<<<dim3((nw + 255) / 256), blk, 0, stream>>>(Wv, Wvb, nw);
  k_cvt<<<dim3((nw + 255) / 256), blk, 0, stream>>>(Wg, Wgb, nw);
  k_cvt<<<dim3((nw + 255) / 256), blk, 0, stream>>>(Wo, Wob, nw);
  k_xT<<<dim3(BFR * 64), blk, 0, stream>>>(x, xbT);

  const size_t lds1 = 4096 * 4 + (4096 * 2 + 3 * DHEAD * DIMC) * 2;
  k1_kv<<<dim3(BFR * HEADS), blk, lds1, stream>>>(xbT, Wkb, Wvb, kv, ksum);

  const size_t lds2 = (2 * 4096 + 3 * 64) * 4 + (2 * 4096 + 3 * DHEAD * DIMC) * 2;
  k2_hid<<<dim3(BFR * HEADS), blk, lds2, stream>>>(xbT, Wqb, Wgb, bg, kv, ksum, hidT);

  const size_t lds3 = (HW * HIDDEN + DIMC * 32) * 2;
  k3_out<<<dim3(BFR * 64), blk, lds3, stream>>>(hidT, Wob, bo, out);
}
